// Attention_34926674051310
// MI455X (gfx1250) — matrix operand order repaired and checked by reading, not compiled, not run on hardware
//
#include <hip/hip_runtime.h>

typedef _Float16 half_t;
typedef __attribute__((ext_vector_type(16))) _Float16 v16h;
typedef __attribute__((ext_vector_type(8)))  _Float16 v8h;
typedef __attribute__((ext_vector_type(8)))  float    v8f;

#define DIM    1024
#define HEADS  8
#define DHEAD  64
#define INNER  512
#define SEQ    2048
#define BATCH  4
#define SCALE_F 8.0f

static __device__ __forceinline__ v8f wmma16x16x32(v16h a, v16h b, v8f c) {
  // D = A(16x32 f16) * B(32x16 f16) + C(16x16 f32)
  return __builtin_amdgcn_wmma_f32_16x16x32_f16(false, a, false, b, (short)0, c,
                                                false, false);
}

// A-fragment (16x32, f16) from row-major f16 memory, leading dim `ld` (halves).
// Lane l: row m = l&15; k = (l>>4)*16 .. +15 contiguous.
static __device__ __forceinline__ v16h a_frag_f16(const half_t* base, int ld, int lane) {
  int m = lane & 15;
  int g = (lane >> 4) * 16;
  const half_t* p = base + (size_t)m * ld;
  v8h lo = *(const v8h*)(p + g);
  v8h hi = *(const v8h*)(p + 8 + g);
  v16h a;
#pragma unroll
  for (int i = 0; i < 8; ++i) { a[i] = lo[i]; a[8 + i] = hi[i]; }
  return a;
}

// B-fragment (32x16, f16) where columns of B are contiguous rows in memory
// (row-major [n][k], leading dim `ld` halves). Lane l: col n = l&15,
// k = (l>>4)*16 .. +15 contiguous.
static __device__ __forceinline__ v16h b_frag_rows(const half_t* base, int ld,
                                                   int koff, int lane) {
  int n = lane & 15;
  int g = (lane >> 4) * 16;
  return *(const v16h*)(base + (size_t)n * ld + koff + g);
}

// Per-lane async global->LDS copy of 16 bytes, tracked by ASYNCcnt.
// LDS address = low 32 bits of the flat shared address (ISA 10.2 aperture rule).
static __device__ __forceinline__ void async_copy_b128(half_t* lds_dst,
                                                       const half_t* gsrc) {
  unsigned int lds_off = (unsigned int)(unsigned long long)(uintptr_t)lds_dst;
  unsigned long long gaddr = (unsigned long long)(uintptr_t)gsrc;
  asm volatile("global_load_async_to_lds_b128 %0, %1, off"
               :: "v"(lds_off), "v"(gaddr) : "memory");
}
static __device__ __forceinline__ void wait_asynccnt0() {
  asm volatile("s_wait_asynccnt 0x0" ::: "memory");
}

// ---------------------------------------------------------------------------
// Kernel 1: QKV projection.  x[8192,1024]f32 @ W[1024,512]f32 -> f16 [b,h,n,d]
// with row l2-normalization for q and k.  z = 0,1,2 selects Wq/Wk/Wv.
// Block: 256 threads (8 waves), wave tile 16(M) x 64(N) = one head slice.
// ---------------------------------------------------------------------------
__global__ __launch_bounds__(256)
void qkv_proj_kernel(const float* __restrict__ x,
                     const float* __restrict__ Wq,
                     const float* __restrict__ Wk,
                     const float* __restrict__ Wv,
                     half_t* __restrict__ Qh,
                     half_t* __restrict__ Kh,
                     half_t* __restrict__ Vh) {
  __shared__ __align__(32) half_t Bt[64 * 32];  // W tile transposed: [n][k] f16

  const int tid  = threadIdx.x;
  const int lane = tid & 31;
  const int wave = tid >> 5;
  const int m0   = blockIdx.x * 128 + wave * 16;  // row in flattened [B*SEQ]
  const int h    = blockIdx.y;                    // head == 64-col tile
  const int n0   = h * DHEAD;
  const int z    = blockIdx.z;

  const float* W = (z == 0) ? Wq : (z == 1) ? Wk : Wv;
  half_t* Dst    = (z == 0) ? Qh : (z == 1) ? Kh : Vh;
  const bool do_norm = (z < 2);

  v8f acc[4] = {};
  for (int k0 = 0; k0 < DIM; k0 += 32) {
    {  // stage W[k0..k0+31][n0..n0+63] transposed into Bt[n][k] as f16
      int k  = tid >> 3;         // 0..31
      int nb = (tid & 7) * 8;    // 0..56
      const float* src = W + (size_t)(k0 + k) * INNER + n0 + nb;
#pragma unroll
      for (int i = 0; i < 8; ++i)
        Bt[(nb + i) * 32 + k] = (half_t)src[i];
    }
    __syncthreads();

    // A fragment: x rows (fp32 -> f16 on the fly); rows are L2-resident.
    v16h a;
    {
      int m = lane & 15;
      int g = (lane >> 4) * 16;
      const float* p = x + (size_t)(m0 + m) * DIM + k0;
#pragma unroll
      for (int i = 0; i < 8; ++i) a[i] = (half_t)p[g + i];
#pragma unroll
      for (int i = 0; i < 8; ++i) a[8 + i] = (half_t)p[8 + g + i];
    }

#pragma unroll
    for (int c = 0; c < 4; ++c) {
      int n = lane & 15;
      int g = (lane >> 4) * 16;
      v16h b = *(const v16h*)(&Bt[(c * 16 + n) * 32 + g]);
      acc[c] = wmma16x16x32(a, b, acc[c]);
    }
    __syncthreads();
  }

  // Row l2 norm (row of 64 lives across the wave's 4 C tiles; reduce over the
  // 16 lanes of each half — xor masks 1/2/4/8 stay inside a half).
  float inv[8];
#pragma unroll
  for (int r = 0; r < 8; ++r) {
    float ss = 0.f;
#pragma unroll
    for (int c = 0; c < 4; ++c) ss += acc[c][r] * acc[c][r];
    ss += __shfl_xor(ss, 1, 32);
    ss += __shfl_xor(ss, 2, 32);
    ss += __shfl_xor(ss, 4, 32);
    ss += __shfl_xor(ss, 8, 32);
    inv[r] = do_norm ? (1.0f / fmaxf(sqrtf(ss), 1e-12f)) : 1.0f;
  }

  // Write f16 to [b][h][n][d]
#pragma unroll
  for (int c = 0; c < 4; ++c) {
#pragma unroll
    for (int r = 0; r < 8; ++r) {
      int m    = m0 + r + ((lane >> 4) << 3);
      int b    = m >> 11;            // /SEQ
      int nrow = m & (SEQ - 1);
      int d    = c * 16 + (lane & 15);
      Dst[(((size_t)(b * HEADS + h)) * SEQ + nrow) * DHEAD + d] =
          (half_t)(acc[c][r] * inv[r]);
    }
  }
}

// ---------------------------------------------------------------------------
// Kernel 2: causal flash attention per (b,h).  Block: 128 threads (4 waves),
// each wave one 16-row Q tile; KV tiles of 64 keys.
//  - K tile (contiguous 8KB) staged via global_load_async_to_lds_b128
//    (ASYNCcnt) — no transpose needed, rows already match B-fragment order.
//  - V tile staged transposed into LDS by hand (needed for the PV B-frags).
//  - P re-laid out C->A fragment through per-wave LDS.
// ---------------------------------------------------------------------------
__global__ __launch_bounds__(128)
void attn_kernel(const half_t* __restrict__ Qh,
                 const half_t* __restrict__ Kh,
                 const half_t* __restrict__ Vh,
                 half_t* __restrict__ Obuf) {
  __shared__ __align__(32) half_t Klds[64 * 64];    // K tile [j][d] (async)
  __shared__ __align__(32) half_t Vt[64 * 64];      // V tile transposed [d][j]
  __shared__ __align__(32) half_t Pl[4][16 * 64];   // per-wave P tile [m][j]

  const int tid  = threadIdx.x;
  const int lane = tid & 31;
  const int wave = tid >> 5;
  const int bh   = blockIdx.y;  // b*HEADS + h
  const int b    = bh >> 3;
  const int h    = bh & 7;
  const int q0   = blockIdx.x * 64 + wave * 16;

  const half_t* Qb = Qh + ((size_t)bh * SEQ + q0) * DHEAD;
  const half_t* Kb = Kh + (size_t)bh * SEQ * DHEAD;
  const half_t* Vb = Vh + (size_t)bh * SEQ * DHEAD;

  // Q fragments for d=0..31 and d=32..63 (reused every KV tile)
  const v16h qa0 = a_frag_f16(Qb, DHEAD, lane);
  const v16h qa1 = a_frag_f16(Qb + 32, DHEAD, lane);

  v8f o[4] = {};
  float mrow[8], lrow[8];
#pragma unroll
  for (int r = 0; r < 8; ++r) { mrow[r] = -1e30f; lrow[r] = 0.f; }

  const int ntiles = blockIdx.x + 1;  // 64-key tiles up to block max row
  for (int t = 0; t < ntiles; ++t) {
    const int j0 = t * 64;

    {  // async-stage K[j0..j0+63][0..63] (contiguous 8KB) into LDS
      const half_t* src = Kb + (size_t)j0 * DHEAD;
#pragma unroll
      for (int i = 0; i < 4; ++i) {
        int e = tid + i * 128;  // 512 chunks of 8 halves
        async_copy_b128(&Klds[e * 8], src + e * 8);
      }
    }
    {  // stage V[j0..j0+63][0..63] transposed into Vt[d][j]
      int jj = tid >> 1;        // 0..63
      int db = (tid & 1) * 32;  // 0, 32
      const half_t* src = Vb + (size_t)(j0 + jj) * DHEAD + db;
#pragma unroll
      for (int i = 0; i < 32; ++i)
        Vt[(db + i) * 64 + jj] = src[i];
    }
    wait_asynccnt0();
    __syncthreads();

    if (j0 <= q0 + 15) {  // tile not fully masked for this wave
      // S = Q K^T over 4 key chunks of 16
      v8f s[4];
#pragma unroll
      for (int cc = 0; cc < 4; ++cc) {
        const half_t* kbase = &Klds[cc * 16 * 64];
        v8f acc = {};
        acc = wmma16x16x32(qa0, b_frag_rows(kbase, 64, 0, lane), acc);
        acc = wmma16x16x32(qa1, b_frag_rows(kbase, 64, 32, lane), acc);
        s[cc] = acc;
      }

      // scale + causal mask
#pragma unroll
      for (int cc = 0; cc < 4; ++cc) {
        int jc = j0 + cc * 16 + (lane & 15);
#pragma unroll
        for (int r = 0; r < 8; ++r) {
          int i = q0 + r + ((lane >> 4) << 3);
          float v = s[cc][r] * SCALE_F;
          if (jc > i) v = -1e30f;
          s[cc][r] = v;
        }
      }

      half_t* P = &Pl[wave][0];
      float alpha[8];
#pragma unroll
      for (int r = 0; r < 8; ++r) {
        float mx = fmaxf(fmaxf(s[0][r], s[1][r]), fmaxf(s[2][r], s[3][r]));
        mx = fmaxf(mx, __shfl_xor(mx, 1, 32));
        mx = fmaxf(mx, __shfl_xor(mx, 2, 32));
        mx = fmaxf(mx, __shfl_xor(mx, 4, 32));
        mx = fmaxf(mx, __shfl_xor(mx, 8, 32));
        float mnew = fmaxf(mrow[r], mx);
        float a = __expf(mrow[r] - mnew);
        float rs = 0.f;
        int m = r + ((lane >> 4) << 3);
#pragma unroll
        for (int cc = 0; cc < 4; ++cc) {
          float p = __expf(s[cc][r] - mnew);
          rs += p;
          P[m * 64 + cc * 16 + (lane & 15)] = (half_t)p;
        }
        rs += __shfl_xor(rs, 1, 32);
        rs += __shfl_xor(rs, 2, 32);
        rs += __shfl_xor(rs, 4, 32);
        rs += __shfl_xor(rs, 8, 32);
        lrow[r] = lrow[r] * a + rs;
        mrow[r] = mnew;
        alpha[r] = a;
      }

#pragma unroll
      for (int c = 0; c < 4; ++c)
#pragma unroll
        for (int r = 0; r < 8; ++r) o[c][r] *= alpha[r];

      // P (C layout) -> A fragments via per-wave LDS round-trip
      v16h pa0 = a_frag_f16(P, 64, lane);       // keys j0..j0+31
      v16h pa1 = a_frag_f16(P + 32, 64, lane);  // keys j0+32..j0+63
#pragma unroll
      for (int c = 0; c < 4; ++c) {
        int n = lane & 15;
        int g = (lane >> 4) * 16;
        v16h vb0 = *(const v16h*)(&Vt[(c * 16 + n) * 64 + g]);
        v16h vb1 = *(const v16h*)(&Vt[(c * 16 + n) * 64 + 32 + g]);
        o[c] = wmma16x16x32(pa0, vb0, o[c]);
        o[c] = wmma16x16x32(pa1, vb1, o[c]);
      }
    }
    __syncthreads();  // protect Klds/Vt before next tile's staging
  }

  // finalize: O /= l, write f16 to merged-head layout [b*SEQ+n][INNER]
#pragma unroll
  for (int c = 0; c < 4; ++c)
#pragma unroll
    for (int r = 0; r < 8; ++r) {
      int i = q0 + r + ((lane >> 4) << 3);
      size_t row = (size_t)b * SEQ + i;
      int col = h * DHEAD + c * 16 + (lane & 15);
      Obuf[row * INNER + col] = (half_t)(o[c][r] / lrow[r]);
    }
}

// ---------------------------------------------------------------------------
// Kernel 3: output projection.  O[8192,512]f16 @ Wo[512,1024]f32 -> f32 out.
// ---------------------------------------------------------------------------
__global__ __launch_bounds__(256)
void oproj_kernel(const half_t* __restrict__ Obuf,
                  const float* __restrict__ Wo,
                  float* __restrict__ out) {
  __shared__ __align__(32) half_t Bt[64 * 32];

  const int tid  = threadIdx.x;
  const int lane = tid & 31;
  const int wave = tid >> 5;
  const int m0   = blockIdx.x * 128 + wave * 16;
  const int n0   = blockIdx.y * 64;

  v8f acc[4] = {};
  for (int k0 = 0; k0 < INNER; k0 += 32) {
    {
      int k  = tid >> 3;
      int nb = (tid & 7) * 8;
      const float* src = Wo + (size_t)(k0 + k) * DIM + n0 + nb;
#pragma unroll
      for (int i = 0; i < 8; ++i)
        Bt[(nb + i) * 32 + k] = (half_t)src[i];
    }
    __syncthreads();

    v16h a = a_frag_f16(Obuf + (size_t)m0 * INNER + k0, INNER, lane);
#pragma unroll
    for (int c = 0; c < 4; ++c) {
      v16h bfr = *(const v16h*)(&Bt[(c * 16 + (lane & 15)) * 32 + ((lane >> 4) * 16)]);
      acc[c] = wmma16x16x32(a, bfr, acc[c]);
    }
    __syncthreads();
  }

#pragma unroll
  for (int c = 0; c < 4; ++c)
#pragma unroll
    for (int r = 0; r < 8; ++r) {
      int m = m0 + r + ((lane >> 4) << 3);
      out[(size_t)m * DIM + n0 + c * 16 + (lane & 15)] = acc[c][r];
    }
}

// ---------------------------------------------------------------------------
extern "C" void kernel_launch(void* const* d_in, const int* in_sizes, int n_in,
                              void* d_out, int out_size, void* d_ws, size_t ws_size,
                              hipStream_t stream) {
  (void)in_sizes; (void)n_in; (void)out_size; (void)ws_size;
  const float* x  = (const float*)d_in[0];
  const float* Wq = (const float*)d_in[1];
  const float* Wk = (const float*)d_in[2];
  const float* Wv = (const float*)d_in[3];
  const float* Wo = (const float*)d_in[4];
  float* out = (float*)d_out;

  const size_t qkv_elems = (size_t)BATCH * HEADS * SEQ * DHEAD;  // 4,194,304
  half_t* Qh = (half_t*)d_ws;
  half_t* Kh = Qh + qkv_elems;
  half_t* Vh = Kh + qkv_elems;
  half_t* Ob = Vh + qkv_elems;  // [B*SEQ][INNER] f16 (8 MB); total ws use 32 MB

  dim3 g1(BATCH * SEQ / 128, HEADS, 3);
  qkv_proj_kernel<<<g1, 256, 0, stream>>>(x, Wq, Wk, Wv, Qh, Kh, Vh);

  dim3 g2(SEQ / 64, BATCH * HEADS);
  attn_kernel<<<g2, 128, 0, stream>>>(Qh, Kh, Vh, Ob);

  dim3 g3(BATCH * SEQ / 128, DIM / 64);
  oproj_kernel<<<g3, 256, 0, stream>>>(Ob, Wo, out);
}
